// MultiHeadDiffAttention_84318797955421
// MI455X (gfx1250) — matrix operand order repaired and checked by reading, not compiled, not run on hardware
//
#include <hip/hip_runtime.h>
#include <hip/hip_bf16.h>

typedef __attribute__((ext_vector_type(16))) __bf16 v16bf;
typedef __attribute__((ext_vector_type(8)))  float  v8f;
typedef __attribute__((ext_vector_type(4)))  unsigned int u32x4;
typedef __attribute__((ext_vector_type(8)))  int i32x8;
typedef __attribute__((ext_vector_type(4)))  int i32x4;

#define DEVI static __device__ __forceinline__

// ---- CDNA5 TDM availability (probe-guarded; manual staging fallback) ----
#if defined(__has_builtin)
#  if __has_builtin(__builtin_amdgcn_tensor_load_to_lds)
#    define HAVE_TDM 1
#  endif
#endif
#ifndef HAVE_TDM
#  define HAVE_TDM 0
#endif

#if defined(__has_builtin)
#  if __has_builtin(__builtin_amdgcn_s_wait_tensorcnt)
#    define WAIT_TENSORCNT() __builtin_amdgcn_s_wait_tensorcnt(0)
#  endif
#endif
#ifndef WAIT_TENSORCNT
#  define WAIT_TENSORCNT() asm volatile("s_wait_tensorcnt 0" ::: "memory")
#endif

namespace {

constexpr int Bc   = 2;
constexpr int Tc   = 2048;
constexpr int Cc   = 1024;
constexpr int Hc   = 16;
constexpr int HSc  = 64;
constexpr int HS2c = 128;
constexpr float LAMBDA_INIT = 0.8f - 0.6f * 0.7408182206817179f;  // 0.8 - 0.6*exp(-0.3)
constexpr float EPSc = 1e-12f;

struct U256 { uint4 lo, hi; };

DEVI v16bf make_frag(uint4 lo, uint4 hi) {
  U256 u{lo, hi};
  return __builtin_bit_cast(v16bf, u);
}

DEVI v8f zero8() {
  v8f z;
#pragma unroll
  for (int i = 0; i < 8; ++i) z[i] = 0.f;
  return z;
}

DEVI v8f wmma_bf16(v16bf a, v16bf b, v8f c) {
  // D = A(16x32 bf16) * B(32x16 bf16) + C(16x16 f32)
  return __builtin_amdgcn_wmma_f32_16x16x32_bf16(false, a, false, b, (short)0, c, false, false);
}

// ---------------------------------------------------------------------------
// Generic WMMA GEMM: out[M,N] = A[M,K] @ W[K,N] + bias
//   ABF16=false : A is fp32, converted to bf16 while staging
//   ABF16=true  : A is bf16 already
//   OUT_HEAD=true : write bf16 to head layout [B,H,T,hd] (n = h*hd + d, m = b*T + t)
//   OUT_HEAD=false: write f32 plain [M,N]
// Block: 256 threads (8 waves). Tile: 128(M) x 64(N), K-step 32.
// ---------------------------------------------------------------------------
template <bool ABF16, bool OUT_HEAD>
__global__ __launch_bounds__(256)
void gemm_wmma_kernel(const void* __restrict__ Ap, const float* __restrict__ W,
                      const float* __restrict__ bias, void* __restrict__ Outp,
                      int M, int N, int K, int hd) {
  __shared__ __bf16 As[128 * 32];   // [m][k] 8 KB
  __shared__ __bf16 BsT[64 * 32];   // [n][k] transposed, 4 KB

  const int tid  = threadIdx.x;
  const int lane = tid & 31;
  const int wave = tid >> 5;
  const int row16  = lane & 15;
  const int halfhi = lane >> 4;
  const int m0 = blockIdx.y * 128;
  const int n0 = blockIdx.x * 64;

  v8f acc[4];
#pragma unroll
  for (int nt = 0; nt < 4; ++nt) acc[nt] = zero8();

  for (int k0 = 0; k0 < K; k0 += 32) {
    __syncthreads();
    // ---- stage A tile (128 x 32) ----
    {
      const int row = tid >> 1;
      const int hsel = (tid & 1) * 16;
      const size_t gofs = (size_t)(m0 + row) * (size_t)K + (size_t)(k0 + hsel);
      __bf16* dst = &As[row * 32 + hsel];
      if (ABF16) {
        const __bf16* A = (const __bf16*)Ap;
        *(uint4*)(dst)     = *(const uint4*)(A + gofs);
        *(uint4*)(dst + 8) = *(const uint4*)(A + gofs + 8);
        if (k0 + 32 < K) __builtin_prefetch((const void*)(A + gofs + 32), 0, 1);
      } else {
        const float* A = (const float*)Ap;
#pragma unroll
        for (int i = 0; i < 16; ++i) dst[i] = (__bf16)A[gofs + i];
        if (k0 + 32 < K) __builtin_prefetch((const void*)(A + gofs + 32), 0, 1);
      }
    }
    // ---- stage B tile transposed (32 x 64 -> BsT[n][k]) ----
    {
      const int r  = tid >> 3;          // k row 0..31
      const int c8 = (tid & 7) * 8;     // col base
      const float* src = W + (size_t)(k0 + r) * (size_t)N + (size_t)(n0 + c8);
#pragma unroll
      for (int i = 0; i < 8; ++i) BsT[(c8 + i) * 32 + r] = (__bf16)src[i];
      if (k0 + 32 < K) __builtin_prefetch((const void*)(src + 32 * (size_t)N), 0, 1);
    }
    __syncthreads();

    // ---- A fragment (16x32), wave-private rows ----
    const __bf16* ap = &As[(wave * 16 + row16) * 32 + halfhi * 8];
    v16bf afrag = make_frag(*(const uint4*)ap, *(const uint4*)(ap + 16));
    const int kb = halfhi * 8;
#pragma unroll
    for (int nt = 0; nt < 4; ++nt) {
      const __bf16* bp = &BsT[(nt * 16 + row16) * 32 + kb];
      v16bf bfrag = make_frag(*(const uint4*)bp, *(const uint4*)(bp + 16));
      acc[nt] = wmma_bf16(afrag, bfrag, acc[nt]);
    }
  }

  // ---- epilogue ----
  const int rbase = halfhi * 8;
#pragma unroll
  for (int nt = 0; nt < 4; ++nt) {
    const int n = n0 + nt * 16 + row16;
    const float bsv = bias[n];
#pragma unroll
    for (int r = 0; r < 8; ++r) {
      const int m = m0 + wave * 16 + rbase + r;
      const float v = acc[nt][r] + bsv;
      if (OUT_HEAD) {
        const int bi = m / Tc, t = m % Tc;
        const int h = n / hd, d = n % hd;
        ((__bf16*)Outp)[((size_t)(bi * Hc + h) * Tc + t) * hd + d] = (__bf16)v;
      } else {
        ((float*)Outp)[(size_t)m * (size_t)N + n] = v;
      }
    }
  }
}

// ---------------------------------------------------------------------------
// Fused dual-stream flash attention + per-head LayerNorm.
// Grid: B*H*(T/128) blocks of 256 threads; each wave owns a 16-row q tile.
// K tiles arrive via the Tensor Data Mover (TDM) when available.
// ---------------------------------------------------------------------------
__global__ __launch_bounds__(256)
void diff_attn_kernel(const __bf16* __restrict__ q1h, const __bf16* __restrict__ q2h,
                      const __bf16* __restrict__ k1h, const __bf16* __restrict__ k2h,
                      const __bf16* __restrict__ vvh, const int* __restrict__ maskp,
                      const float* __restrict__ lq1, const float* __restrict__ lk1,
                      const float* __restrict__ lq2, const float* __restrict__ lk2,
                      const float* __restrict__ ln_w, const float* __restrict__ ln_b,
                      __bf16* __restrict__ ybuf) {
  __shared__ __bf16 Ks1[32 * 64];        // [key][hs]   4 KB
  __shared__ __bf16 Ks2[32 * 64];        //             4 KB
  __shared__ __bf16 VT[128 * 32];        // [n][key]    8 KB (transposed)
  __shared__ __bf16 Pst[8][16 * 32];     // per-wave P  8 KB

  const int tid  = threadIdx.x;
  const int lane = tid & 31;
  const int wave = tid >> 5;
  const int row16  = lane & 15;
  const int halfhi = lane >> 4;

  const int blk = blockIdx.x;
  const int qc  = blk % (Tc / 128);
  const int bh  = blk / (Tc / 128);
  const int h   = bh % Hc;
  const int bi  = bh / Hc;
  const size_t base64  = (size_t)bh * Tc * HSc;
  const size_t base128 = (size_t)bh * Tc * HS2c;
  const int q0 = qc * 128 + wave * 16;

#if HAVE_TDM
  // D# group1 for a 32x64 bf16 tile (data_size=2B, tensor_dim0=64, tensor_dim1=32,
  // tile_dim0=64, tile_dim1=32, tensor_dim0_stride=64), per ISA 8.4 bit packing.
  typedef __attribute__((address_space(3))) __bf16 bf16_as3;
  const unsigned ldsK1 = (unsigned)(unsigned long long)(bf16_as3*)&Ks1[0];
  const unsigned ldsK2 = (unsigned)(unsigned long long)(bf16_as3*)&Ks2[0];
  const i32x8 g1K = { (int)(1u << 16),   // data_size = 1 (2 bytes)
                      (int)(64u << 16),  // tensor_dim0[15:0]
                      (int)(32u << 16),  // tensor_dim0[31:16]=0 | tensor_dim1[15:0]
                      (int)(64u << 16),  // tensor_dim1[31:16]=0 | tile_dim0
                      32,                // tile_dim1 | tile_dim2=0
                      64,                // tensor_dim0_stride[31:0]
                      0, 0 };
  const i32x4 z4 = {0, 0, 0, 0};
#if __clang_major__ >= 23
  const i32x8 z8 = {0, 0, 0, 0, 0, 0, 0, 0};
#endif
#endif

  // lambda = exp(lq1.lk1) - exp(lq2.lk2) + lambda_init (per head)
  float lam;
  {
    float s1 = 0.f, s2 = 0.f;
#pragma unroll 8
    for (int i = 0; i < HSc; ++i) {
      s1 += lq1[h * HSc + i] * lk1[h * HSc + i];
      s2 += lq2[h * HSc + i] * lk2[h * HSc + i];
    }
    lam = __expf(s1) - __expf(s2) + LAMBDA_INIT;
  }

  // Q fragments (16x64 per stream -> 2 k-frags each)
  v16bf qf1[2], qf2[2];
  {
    const __bf16* q1p = q1h + base64 + (size_t)(q0 + row16) * HSc;
    const __bf16* q2p = q2h + base64 + (size_t)(q0 + row16) * HSc;
#pragma unroll
    for (int f = 0; f < 2; ++f) {
      const int off = f * 32 + halfhi * 8;
      qf1[f] = make_frag(*(const uint4*)(q1p + off), *(const uint4*)(q1p + off + 16));
      qf2[f] = make_frag(*(const uint4*)(q2p + off), *(const uint4*)(q2p + off + 16));
    }
  }

  bool rowmask[8];
#pragma unroll
  for (int r = 0; r < 8; ++r)
    rowmask[r] = (maskp[bi * Tc + q0 + halfhi * 8 + r] != 0);

  v8f O1[8], O2[8];
  float m1[8], l1[8], m2[8], l2[8];
#pragma unroll
  for (int nt = 0; nt < 8; ++nt) { O1[nt] = zero8(); O2[nt] = zero8(); }
#pragma unroll
  for (int r = 0; r < 8; ++r) { m1[r] = m2[r] = -3.0e38f; l1[r] = l2[r] = 0.f; }

  auto stream_step = [&](const v16bf (&qf)[2], const __bf16* __restrict__ Ks,
                         float (&mm)[8], float (&ll)[8], v8f (&O)[8]) {
    // S = Q K^T : 16 rows x 32 keys (two 16x16 tiles), k-dim 64 = 2 frags
    v8f S0 = zero8(), S1 = zero8();
#pragma unroll
    for (int f = 0; f < 2; ++f) {
      const __bf16* b0 = &Ks[(row16)*64 + f * 32 + halfhi * 8];
      const __bf16* b1 = &Ks[(16 + row16) * 64 + f * 32 + halfhi * 8];
      v16bf bf0 = make_frag(*(const uint4*)b0, *(const uint4*)(b0 + 16));
      v16bf bf1 = make_frag(*(const uint4*)b1, *(const uint4*)(b1 + 16));
      S0 = wmma_bf16(qf[f], bf0, S0);
      S1 = wmma_bf16(qf[f], bf1, S1);
    }
    float P0[8], P1[8];
#pragma unroll
    for (int r = 0; r < 8; ++r) {
      P0[r] = rowmask[r] ? S0[r] * 0.125f : 0.f;   // masked row -> uniform softmax
      P1[r] = rowmask[r] ? S1[r] * 0.125f : 0.f;
    }
    // online softmax per row (row lives across 16 lanes of one half)
#pragma unroll
    for (int r = 0; r < 8; ++r) {
      float mx = fmaxf(P0[r], P1[r]);
#pragma unroll
      for (int off = 1; off < 16; off <<= 1) mx = fmaxf(mx, __shfl_xor(mx, off));
      const float mn = fmaxf(mm[r], mx);
      const float alpha = __expf(mm[r] - mn);
      P0[r] = __expf(P0[r] - mn);
      P1[r] = __expf(P1[r] - mn);
      float rs = P0[r] + P1[r];
#pragma unroll
      for (int off = 1; off < 16; off <<= 1) rs += __shfl_xor(rs, off);
      ll[r] = ll[r] * alpha + rs;
      mm[r] = mn;
#pragma unroll
      for (int nt = 0; nt < 8; ++nt) O[nt][r] *= alpha;
      Pst[wave][(halfhi * 8 + r) * 32 + row16]      = (__bf16)P0[r];
      Pst[wave][(halfhi * 8 + r) * 32 + 16 + row16] = (__bf16)P1[r];
    }
    asm volatile("s_wait_dscnt 0" ::: "memory");   // wave-local P visible
    // P (16x32) as A-fragment, V (32x128) from transposed LDS tile
    const __bf16* pp = &Pst[wave][row16 * 32 + halfhi * 8];
    v16bf pf = make_frag(*(const uint4*)pp, *(const uint4*)(pp + 16));
#pragma unroll
    for (int nt = 0; nt < 8; ++nt) {
      const __bf16* vp = &VT[(nt * 16 + row16) * 32 + halfhi * 8];
      v16bf vf = make_frag(*(const uint4*)vp, *(const uint4*)(vp + 16));
      O[nt] = wmma_bf16(pf, vf, O[nt]);
    }
  };

  for (int kb = 0; kb < Tc; kb += 32) {
    __syncthreads();
#if HAVE_TDM
    // ---- TDM: DMA both K tiles (32x64 bf16) global -> LDS, wave 0 issues ----
    if (wave == 0) {
      const unsigned long long ga1 =
          (unsigned long long)(k1h + base64 + (size_t)kb * HSc);
      const unsigned long long ga2 =
          (unsigned long long)(k2h + base64 + (size_t)kb * HSc);
      u32x4 g0a = { 1u, ldsK1, (unsigned)(ga1 & 0xffffffffu),
                    (unsigned)((ga1 >> 32) & 0x01ffffffu) | 0x80000000u };
      u32x4 g0b = { 1u, ldsK2, (unsigned)(ga2 & 0xffffffffu),
                    (unsigned)((ga2 >> 32) & 0x01ffffffu) | 0x80000000u };
#if __clang_major__ >= 23
      __builtin_amdgcn_tensor_load_to_lds(g0a, g1K, z4, z4, z8, 0);
      __builtin_amdgcn_tensor_load_to_lds(g0b, g1K, z4, z4, z8, 0);
#else
      __builtin_amdgcn_tensor_load_to_lds(g0a, g1K, z4, z4, 0);
      __builtin_amdgcn_tensor_load_to_lds(g0b, g1K, z4, z4, 0);
#endif
      WAIT_TENSORCNT();
    }
#else
    // ---- manual K staging (32x64 each): 8 contiguous halfs per thread ----
    {
      const int idx  = tid * 8;
      const int krow = idx >> 6;
      const int kcol = idx & 63;
      const size_t go = base64 + (size_t)(kb + krow) * HSc + kcol;
      *(uint4*)(&Ks1[idx]) = *(const uint4*)(k1h + go);
      *(uint4*)(&Ks2[idx]) = *(const uint4*)(k2h + go);
      if (kb + 32 < Tc) {
        __builtin_prefetch((const void*)(k1h + go + 32 * HSc), 0, 1);
        __builtin_prefetch((const void*)(k2h + go + 32 * HSc), 0, 1);
      }
    }
#endif
    // ---- stage V transposed: 16 contiguous halfs/thread -> VT[n][key] ----
    {
      const int idx2 = tid * 16;
      const int vkey = idx2 >> 7;
      const int vn   = idx2 & 127;
      const __bf16* sv = vvh + base128 + (size_t)(kb + vkey) * HS2c + vn;
      v16bf tv = make_frag(*(const uint4*)sv, *(const uint4*)(sv + 8));
#pragma unroll
      for (int i = 0; i < 16; ++i) VT[(vn + i) * 32 + vkey] = tv[i];
      if (kb + 32 < Tc)
        __builtin_prefetch((const void*)(sv + 32 * HS2c), 0, 1);
    }
    __syncthreads();

    stream_step(qf1, Ks1, m1, l1, O1);
    stream_step(qf2, Ks2, m2, l2, O2);
  }

  // ---- finalize: y = O1/l1 - lam*O2/l2, per-row LayerNorm over 128, scale ----
#pragma unroll
  for (int r = 0; r < 8; ++r) {
    const float i1 = 1.f / l1[r];
    const float i2 = lam / l2[r];
#pragma unroll
    for (int nt = 0; nt < 8; ++nt) O1[nt][r] = O1[nt][r] * i1 - O2[nt][r] * i2;
  }
#pragma unroll
  for (int r = 0; r < 8; ++r) {
    float s = 0.f;
#pragma unroll
    for (int nt = 0; nt < 8; ++nt) s += O1[nt][r];
#pragma unroll
    for (int off = 1; off < 16; off <<= 1) s += __shfl_xor(s, off);
    const float mean = s * (1.f / 128.f);
    float vs = 0.f;
#pragma unroll
    for (int nt = 0; nt < 8; ++nt) { const float d = O1[nt][r] - mean; vs += d * d; }
#pragma unroll
    for (int off = 1; off < 16; off <<= 1) vs += __shfl_xor(vs, off);
    const float inv = rsqrtf(vs * (1.f / 128.f) + EPSc);
#pragma unroll
    for (int nt = 0; nt < 8; ++nt) O1[nt][r] = (O1[nt][r] - mean) * inv;
  }
  const float c1 = 1.f - LAMBDA_INIT;
#pragma unroll
  for (int nt = 0; nt < 8; ++nt) {
    const int n = nt * 16 + row16;
    const float w = ln_w[n], bb = ln_b[n];
#pragma unroll
    for (int r = 0; r < 8; ++r) {
      const int t = q0 + halfhi * 8 + r;
      const float v = (O1[nt][r] * w + bb) * c1;
      ybuf[(size_t)(bi * Tc + t) * (2 * Cc) + h * HS2c + n] = (__bf16)v;
    }
  }
}

}  // namespace

extern "C" void kernel_launch(void* const* d_in, const int* in_sizes, int n_in,
                              void* d_out, int out_size, void* d_ws, size_t ws_size,
                              hipStream_t stream) {
  const float* q    = (const float*)d_in[0];
  const float* kin  = (const float*)d_in[1];
  const float* vin  = (const float*)d_in[2];
  const int*   mask = (const int*)d_in[3];
  const float* Wq1 = (const float*)d_in[4];  const float* bq1 = (const float*)d_in[5];
  const float* Wq2 = (const float*)d_in[6];  const float* bq2 = (const float*)d_in[7];
  const float* Wk1 = (const float*)d_in[8];  const float* bk1 = (const float*)d_in[9];
  const float* Wk2 = (const float*)d_in[10]; const float* bk2 = (const float*)d_in[11];
  const float* Wv  = (const float*)d_in[12]; const float* bv  = (const float*)d_in[13];
  const float* Wc  = (const float*)d_in[14]; const float* bc  = (const float*)d_in[15];
  const float* ln_w = (const float*)d_in[16]; const float* ln_b = (const float*)d_in[17];
  const float* lq1 = (const float*)d_in[18]; const float* lk1 = (const float*)d_in[19];
  const float* lq2 = (const float*)d_in[20]; const float* lk2 = (const float*)d_in[21];

  const int Mrows = Bc * Tc;                    // 4096
  char* ws = (char*)d_ws;
  const size_t szQK = (size_t)Mrows * Cc * sizeof(__bf16);       // 8 MB each
  const size_t szV  = (size_t)Mrows * 2 * Cc * sizeof(__bf16);   // 16 MB
  __bf16* q1h  = (__bf16*)(ws);
  __bf16* q2h  = (__bf16*)(ws + szQK);
  __bf16* k1h  = (__bf16*)(ws + 2 * szQK);
  __bf16* k2h  = (__bf16*)(ws + 3 * szQK);
  __bf16* vvh  = (__bf16*)(ws + 4 * szQK);
  __bf16* ybuf = (__bf16*)(ws + 4 * szQK + szV);

  dim3 blk(256);
  dim3 gP(Cc / 64, Mrows / 128);        // 16 x 32
  dim3 gV(2 * Cc / 64, Mrows / 128);    // 32 x 32

  // projections (fp32 in -> bf16 head-layout out)
  gemm_wmma_kernel<false, true><<<gP, blk, 0, stream>>>(q,   Wq1, bq1, q1h, Mrows, Cc,     Cc, HSc);
  gemm_wmma_kernel<false, true><<<gP, blk, 0, stream>>>(q,   Wq2, bq2, q2h, Mrows, Cc,     Cc, HSc);
  gemm_wmma_kernel<false, true><<<gP, blk, 0, stream>>>(kin, Wk1, bk1, k1h, Mrows, Cc,     Cc, HSc);
  gemm_wmma_kernel<false, true><<<gP, blk, 0, stream>>>(kin, Wk2, bk2, k2h, Mrows, Cc,     Cc, HSc);
  gemm_wmma_kernel<false, true><<<gV, blk, 0, stream>>>(vin, Wv,  bv,  vvh, Mrows, 2 * Cc, Cc, HS2c);

  // fused dual-stream flash attention + per-head LN
  diff_attn_kernel<<<dim3(Bc * Hc * (Tc / 128)), blk, 0, stream>>>(
      q1h, q2h, k1h, k2h, vvh, mask, lq1, lk1, lq2, lk2, ln_w, ln_b, ybuf);

  // output projection (bf16 in -> fp32 out)
  gemm_wmma_kernel<true, false><<<gP, blk, 0, stream>>>(ybuf, Wc, bc, (float*)d_out,
                                                        Mrows, Cc, 2 * Cc, 0);
}
